// MaskGenerator_72232759984629
// MI455X (gfx1250) — matrix operand order repaired and checked by reading, not compiled, not run on hardware
//
#include <hip/hip_runtime.h>

typedef __attribute__((ext_vector_type(16))) __bf16 v16bf;
typedef __attribute__((ext_vector_type(8)))  __bf16 v8bf;
typedef __attribute__((ext_vector_type(4)))  __bf16 v4bf;
typedef __attribute__((ext_vector_type(8)))  float  v8f;

#define B_ROWS 16384
#define ZDIM   512
#define HIDDEN 128

// monotone float -> uint mapping (larger float <=> larger uint)
__device__ __forceinline__ unsigned sortkey(float f) {
    unsigned u = __float_as_uint(f);
    return u ^ ((u & 0x80000000u) ? 0xFFFFFFFFu : 0x80000000u);
}

// Load a 16-bf16 A-fragment: 8 contiguous elems at base, 8 at base+16
__device__ __forceinline__ v16bf load_a_frag(const __bf16* base) {
    v8bf lo = *(const v8bf*)base;
    v8bf hi = *(const v8bf*)(base + 16);
    return __builtin_shufflevector(lo, hi, 0, 1, 2, 3, 4, 5, 6, 7,
                                           8, 9, 10, 11, 12, 13, 14, 15);
}

// ---------------------------------------------------------------------------
// Setup: convert a (K x N) fp32 weight matrix into WMMA-B bf16 fragments.
// Fragment for (kt, ct, lane): 16 contiguous bf16 =
//   W[kt*32 + (lane>>4)*8 + (j<8 ? j : j+8)][ct*16 + (lane&15)], j = 0..15
// ---------------------------------------------------------------------------
__global__ __launch_bounds__(256)
void swizzle_w_kernel(const float* __restrict__ W, __bf16* __restrict__ frag,
                      int K, int N)
{
    const int t = blockIdx.x * blockDim.x + threadIdx.x;
    const int total = (K / 32) * (N / 16) * 32;
    if (t >= total) return;
    const int lane = t & 31;
    const int tile = t >> 5;
    const int nt   = N / 16;
    const int ct   = tile % nt;
    const int kt   = tile / nt;
    const int n    = ct * 16 + (lane & 15);
    const int kb   = kt * 32 + (lane >> 4) * 8;
    __bf16* out = frag + (size_t)t * 16;
    const float* src = W + (size_t)kb * N + n;
    #pragma unroll
    for (int j = 0; j < 16; ++j)
        out[j] = (__bf16)src[(size_t)(j < 8 ? j : j + 8) * N];
}

// ---------------------------------------------------------------------------
// Fused head kernel: 16 rows per block; h-GEMM + logits-GEMM (WMMA) entirely
// in LDS, then exact per-row top-k (k = C/2) radix select, write 0/2 mask.
// ---------------------------------------------------------------------------
__global__ __launch_bounds__(256)
void mask_head_kernel(const float* __restrict__ z,       // (B, ZDIM) fp32
                      const __bf16* __restrict__ w1frag, // W1 fragments (this head)
                      const float* __restrict__ b1h,     // (HIDDEN)
                      const __bf16* __restrict__ w2frag, // W2 fragments (this head)
                      const float* __restrict__ b2,      // (C)
                      float* __restrict__ outp,          // (B, C)
                      int C, int keep)
{
    extern __shared__ char smem[];
    float*    logits = (float*)smem;                                     // 16*C fp32
    __bf16*   zt     = (__bf16*)(smem + (size_t)16 * C * 4);             // 16*ZDIM bf16
    __bf16*   hbuf   = zt + 16 * ZDIM;                                   // 16*HIDDEN bf16
    unsigned* hist   = (unsigned*)((char*)(hbuf + 16 * HIDDEN));         // 256 bins
    unsigned* ctrl   = hist + 256;                                       // pfx/rem/tiectr

    const int tid  = threadIdx.x;
    const int wave = tid >> 5;          // 8 wave32s
    const int lane = tid & 31;
    const int m    = lane & 15;
    const int half = lane >> 4;
    const int k0   = half * 8;
    const int row0 = blockIdx.x * 16;

    // ---- Phase 0: stage z tile (16 x ZDIM) into LDS as bf16, once ----------
    {
        const float* zsrc = z + (size_t)row0 * ZDIM;   // 16 contiguous rows
        for (int i = tid * 4; i < 16 * ZDIM; i += 256 * 4) {
            const float4 f = *(const float4*)(zsrc + i);
            v4bf v;
            v[0] = (__bf16)f.x; v[1] = (__bf16)f.y;
            v[2] = (__bf16)f.z; v[3] = (__bf16)f.w;
            *(v4bf*)(zt + i) = v;
        }
    }
    __syncthreads();

    // ---- Phase 1: h = relu(z @ W1 + b1), 16 x HIDDEN -----------------------
    {
        v8f acc = {};
        const int n = wave * 16 + m;    // hidden column
        #pragma unroll 4
        for (int kt = 0; kt < ZDIM / 32; ++kt) {
            v16bf a = load_a_frag(&zt[m * ZDIM + kt * 32 + k0]);
            v16bf b = *(const v16bf*)&w1frag[((size_t)(kt * (HIDDEN / 16) + wave) * 32 + lane) * 16];
            acc = __builtin_amdgcn_wmma_f32_16x16x32_bf16(false, a, false, b,
                                                          (short)0, acc, false, false);
        }
        const float bias = b1h[n];
        #pragma unroll
        for (int r = 0; r < 8; ++r) {
            float v = acc[r] + bias;
            v = v > 0.0f ? v : 0.0f;
            hbuf[(r + half * 8) * HIDDEN + n] = (__bf16)v;
        }
    }
    __syncthreads();

    // ---- Phase 2: logits = h @ W2 + b2, 16 x C -----------------------------
    for (int ct = wave; ct < C / 16; ct += 8) {
        const int n = ct * 16 + m;
        v8f acc = {};
        #pragma unroll
        for (int kt = 0; kt < HIDDEN / 32; ++kt) {
            v16bf a = load_a_frag(&hbuf[m * HIDDEN + kt * 32 + k0]);
            v16bf b = *(const v16bf*)&w2frag[((size_t)(kt * (C / 16) + ct) * 32 + lane) * 16];
            acc = __builtin_amdgcn_wmma_f32_16x16x32_bf16(false, a, false, b,
                                                          (short)0, acc, false, false);
        }
        const float bias = b2[n];
        #pragma unroll
        for (int r = 0; r < 8; ++r)
            logits[(size_t)(r + half * 8) * C + n] = acc[r] + bias;
    }
    __syncthreads();

    // ---- Phase 3: exact top-k threshold per row (MSB-first radix select) ---
    for (int row = 0; row < 16; ++row) {
        const float* lr = logits + (size_t)row * C;
        unsigned pfx = 0, rem = (unsigned)keep;
        for (int pass = 0; pass < 4; ++pass) {
            hist[tid] = 0;                       // 256 threads == 256 bins
            __syncthreads();
            const int shift = 24 - pass * 8;
            for (int c = tid; c < C; c += 256) {
                unsigned u = sortkey(lr[c]);
                bool ok = (pass == 0) || ((u >> (shift + 8)) == pfx);
                if (ok) atomicAdd(&hist[(u >> shift) & 0xFFu], 1u);
            }
            __syncthreads();
            if (tid == 0) {
                unsigned cum = 0;
                int bin = 255;
                for (; bin > 0; --bin) {
                    unsigned h = hist[bin];
                    if (cum + h >= rem) break;
                    cum += h;
                }
                ctrl[0] = (pfx << 8) | (unsigned)bin;
                ctrl[1] = rem - cum;
            }
            __syncthreads();
            pfx = ctrl[0];
            rem = ctrl[1];
            __syncthreads();
        }
        const unsigned T = pfx;                  // exact bits of k-th largest
        if (tid == 0) ctrl[2] = 0;
        __syncthreads();
        float* orow = outp + (size_t)(row0 + row) * C;
        for (int c = tid; c < C; c += 256) {
            unsigned u = sortkey(lr[c]);
            float o = 0.0f;
            if (u > T) {
                o = 2.0f;                        // hard mask / SPARSITY
            } else if (u == T) {
                unsigned i = atomicAdd(&ctrl[2], 1u);
                if (i < rem) o = 2.0f;
            }
            orow[c] = o;
        }
        __syncthreads();
    }
}

extern "C" void kernel_launch(void* const* d_in, const int* in_sizes, int n_in,
                              void* d_out, int out_size, void* d_ws, size_t ws_size,
                              hipStream_t stream) {
    (void)in_sizes; (void)n_in; (void)out_size; (void)ws_size;
    const float* z  = (const float*)d_in[0];
    const float* W1 = (const float*)d_in[1];
    const float* b1 = (const float*)d_in[2];
    const int CH[4] = {512, 1024, 2048, 4096};

    __bf16* ws = (__bf16*)d_ws;

    // ---- workspace layout: W1 fragments (4 heads), then W2 fragments ------
    size_t w1Ofs[4], w2Ofs[4], ofs = 0;
    const size_t w1FragElems = (size_t)(ZDIM / 32) * (HIDDEN / 16) * 32 * 16; // 65536
    for (int i = 0; i < 4; ++i) { w1Ofs[i] = ofs; ofs += w1FragElems; }
    for (int i = 0; i < 4; ++i) { w2Ofs[i] = ofs; ofs += (size_t)HIDDEN * CH[i]; }

    // ---- setup: swizzle weights into WMMA-B bf16 fragments ----------------
    for (int i = 0; i < 4; ++i) {
        const int total1 = (ZDIM / 32) * (HIDDEN / 16) * 32;
        swizzle_w_kernel<<<dim3((total1 + 255) / 256), dim3(256), 0, stream>>>(
            W1 + (size_t)i * ZDIM * HIDDEN, ws + w1Ofs[i], ZDIM, HIDDEN);
        const float* W2 = (const float*)d_in[3 + 2 * i];
        const int total2 = (HIDDEN / 32) * (CH[i] / 16) * 32;
        swizzle_w_kernel<<<dim3((total2 + 255) / 256), dim3(256), 0, stream>>>(
            W2, ws + w2Ofs[i], HIDDEN, CH[i]);
    }

    // ---- fused head kernels ------------------------------------------------
    size_t outOfs = 0;
    for (int i = 0; i < 4; ++i) {
        const float* b2 = (const float*)d_in[4 + 2 * i];
        const int C = CH[i];
        size_t smem = (size_t)16 * C * 4          // logits
                    + (size_t)16 * ZDIM * 2       // z tile bf16
                    + (size_t)16 * HIDDEN * 2     // h tile bf16
                    + 256 * 4 + 16;               // histogram + ctrl
        mask_head_kernel<<<dim3(B_ROWS / 16), dim3(256), smem, stream>>>(
            z, ws + w1Ofs[i], b1 + (size_t)i * HIDDEN,
            ws + w2Ofs[i], b2,
            (float*)d_out + outOfs, C, C / 2);
        outOfs += (size_t)B_ROWS * C;
    }
}
